// GlobalWorkspaceBlock_33681133535775
// MI455X (gfx1250) — matrix operand order repaired and checked by reading, not compiled, not run on hardware
//
#include <hip/hip_runtime.h>
#include <hip/hip_bf16.h>
#include <math.h>

// ---------------------------------------------------------------------------
// Types / helpers
// ---------------------------------------------------------------------------
typedef __attribute__((ext_vector_type(16))) __bf16 v16bf;
typedef __attribute__((ext_vector_type(8)))  float  v8f;

__device__ __forceinline__ __bf16 f2bf(float f) {
  unsigned u = __builtin_bit_cast(unsigned, f);
  unsigned r = (u + 0x7FFFu + ((u >> 16) & 1u)) >> 16;
  unsigned short s = (unsigned short)r;
  return __builtin_bit_cast(__bf16, s);
}

#define D_MODEL 1024
#define N_HEADS 16
#define HEAD_D  64
#define SEQ     1024
#define BATCH   4
#define DFF     4096
#define ROWS    (BATCH * SEQ)   // 4096
#define MAXSTEPS 24

// ---------------------------------------------------------------------------
// CDNA5 async global->LDS copy (ASYNCcnt-tracked), with sync fallback
// ---------------------------------------------------------------------------
#if __has_builtin(__builtin_amdgcn_global_load_async_to_lds_b128)
#define HAS_ASYNC_LDS 1
#else
#define HAS_ASYNC_LDS 0
#endif

typedef int v4i_gcc __attribute__((vector_size(16)));

__device__ __forceinline__ void async_cp16(const __bf16* g, __bf16* l) {
#if HAS_ASYNC_LDS
  __builtin_amdgcn_global_load_async_to_lds_b128(
      (__attribute__((address_space(1))) v4i_gcc*)g,
      (__attribute__((address_space(3))) v4i_gcc*)l, 0, 0);
#else
  *(uint4*)l = *(const uint4*)g;
#endif
}

template <int N>
__device__ __forceinline__ void wait_async() {
#if HAS_ASYNC_LDS
#if __has_builtin(__builtin_amdgcn_s_wait_asynccnt)
  __builtin_amdgcn_s_wait_asynccnt(N);
#else
  asm volatile("s_wait_asynccnt %0" ::"n"(N));
#endif
#endif
}

// ---------------------------------------------------------------------------
// Elementwise prep kernels (fp32 -> bf16)
// ---------------------------------------------------------------------------
// Build concat([x, orig]) as bf16, shape (4096, 2048)
__global__ void pack_cat_kernel(const float* __restrict__ x,
                                const float* __restrict__ o,
                                __bf16* __restrict__ dst) {
  int i = blockIdx.x * 256 + threadIdx.x;       // over 4096*2048
  int row = i >> 11;
  int col = i & 2047;
  float v = (col < D_MODEL) ? x[(size_t)row * D_MODEL + col]
                            : o[(size_t)row * D_MODEL + (col - D_MODEL)];
  dst[i] = f2bf(v);
}

// Convert weight (K x N, f32) -> transposed bf16 (N x K) so GEMM B-tiles are
// straight row-major copies (async-LDS friendly, K contiguous per lane).
__global__ void cvt_bf16_t_kernel(const float* __restrict__ src,
                                  __bf16* __restrict__ dst,
                                  int K, int N) {
  int i = blockIdx.x * 256 + threadIdx.x;   // coalesced over N
  if (i < K * N) {
    int k = i / N, n = i - k * N;
    dst[(size_t)n * K + k] = f2bf(src[i]);
  }
}

// ---------------------------------------------------------------------------
// WMMA bf16 GEMM: C(MxN) = A(MxK) * Bt(NxK)^T, fp32 accumulate.
// Block: 256 threads = 8 waves; tile BM=128, BN=128, BK=32, double-buffered
// LDS with async global->LDS staging. Waves 2x4; each computes 64x32 = 4x2
// WMMA tiles of 16x16.
// ---------------------------------------------------------------------------
#define EPI_GATE 0
#define EPI_QKV  1
#define EPI_BIAS 2
#define EPI_GELU 3

constexpr int BM = 128, BN = 128, BK = 32;
constexpr int LDA = BK + 8;   // padded LDS row stride (bf16 units, keeps 16B align)
constexpr int LDB = BK + 8;

template <int EPI>
__global__ __launch_bounds__(256, 2)
void gemm_bf16_kernel(const __bf16* __restrict__ A,    // M x K
                      const __bf16* __restrict__ Bt,   // N x K (pre-transposed)
                      int M, int N, int K,
                      const float* __restrict__ bias,
                      float*  __restrict__ outF,
                      __bf16* __restrict__ outB,
                      const float* __restrict__ ex0,   // GATE: x
                      const float* __restrict__ ex1,   // GATE: orig
                      const float* __restrict__ ex2,   // GATE: step_embed
                      const int*   __restrict__ exInt) // GATE: think_step
{
  __shared__ __bf16 As[2][BM * LDA];
  __shared__ __bf16 Bs[2][BN * LDB];

  const int tid  = threadIdx.x;
  const int lane = tid & 31;
  const int wave = tid >> 5;
  const int wm   = wave >> 2;   // 0..1
  const int wn   = wave & 3;    // 0..3
  const int hi   = lane >> 4;   // lane half
  const int ln   = lane & 15;

  const int rowBlk = blockIdx.y * BM;
  const int colBlk = blockIdx.x * BN;

  v8f acc[4][2];
#pragma unroll
  for (int rt = 0; rt < 4; ++rt)
#pragma unroll
    for (int ct = 0; ct < 2; ++ct)
#pragma unroll
      for (int j = 0; j < 8; ++j) acc[rt][ct][j] = 0.f;

  // Stage one BMxBK (and BNxBK) tile: 512 chunks of 16B each -> 2 per thread
  // per matrix. Chunk c: row = c>>2, 16B segment = c&3.
  auto issueStage = [&](int buf, int k0) {
#pragma unroll
    for (int i = 0; i < 2; ++i) {
      int c = tid + 256 * i;
      int row = c >> 2, seg = c & 3;
      async_cp16(&A[(size_t)(rowBlk + row) * K + k0 + seg * 8],
                 &As[buf][row * LDA + seg * 8]);
    }
#pragma unroll
    for (int i = 0; i < 2; ++i) {
      int c = tid + 256 * i;
      int n = c >> 2, seg = c & 3;
      async_cp16(&Bt[(size_t)(colBlk + n) * K + k0 + seg * 8],
                 &Bs[buf][n * LDB + seg * 8]);
    }
  };

  const int kTiles = K / BK;
  issueStage(0, 0);
  for (int kt = 0; kt < kTiles; ++kt) {
    const int cur = kt & 1;
    if (kt + 1 < kTiles) {
      issueStage(cur ^ 1, (kt + 1) * BK);
      wait_async<4>();   // previous stage's 4 copies landed; next 4 in flight
    } else {
      wait_async<0>();
    }
    __syncthreads();

    // ---- fragments from LDS ----
    v16bf afr[4], bfr[2];
#pragma unroll
    for (int rt = 0; rt < 4; ++rt) {
      const __bf16* p = &As[cur][(wm * 64 + rt * 16 + ln) * LDA + 16 * hi];
#pragma unroll
      for (int j = 0; j < 8; ++j) afr[rt][j]     = p[j];        // K = 16*hi + j
#pragma unroll
      for (int j = 0; j < 8; ++j) afr[rt][8 + j] = p[8 + j];    // K = 16*hi + 8 + j
    }
#pragma unroll
    for (int ct = 0; ct < 2; ++ct) {
      const __bf16* p = &Bs[cur][(wn * 32 + ct * 16 + ln) * LDB + 16 * hi];
#pragma unroll
      for (int j = 0; j < 16; ++j) bfr[ct][j] = p[j];            // K = 16*hi + j
    }
#pragma unroll
    for (int rt = 0; rt < 4; ++rt)
#pragma unroll
      for (int ct = 0; ct < 2; ++ct)
        acc[rt][ct] = __builtin_amdgcn_wmma_f32_16x16x32_bf16(
            false, afr[rt], false, bfr[ct], (short)0, acc[rt][ct], false, false);
    __syncthreads();   // protect buffer reuse two iterations out
  }

  // ---- epilogue ----
  int clampedStep = 0;
  if constexpr (EPI == EPI_GATE) {
    int st = exInt[0];
    clampedStep = st < (MAXSTEPS - 1) ? st : (MAXSTEPS - 1);
    clampedStep = clampedStep < 0 ? 0 : clampedStep;
  }
#pragma unroll
  for (int rt = 0; rt < 4; ++rt) {
#pragma unroll
    for (int ct = 0; ct < 2; ++ct) {
#pragma unroll
      for (int e = 0; e < 8; ++e) {
        const int m = rowBlk + wm * 64 + rt * 16 + e + 8 * hi;
        const int n = colBlk + wn * 32 + ct * 16 + ln;
        const float v = acc[rt][ct][e];
        if constexpr (EPI == EPI_GATE) {
          float g  = 1.f / (1.f + __expf(-(v + bias[n])));
          float xv = ex0[(size_t)m * D_MODEL + n];
          float ov = ex1[(size_t)m * D_MODEL + n];
          float se = ex2[clampedStep * D_MODEL + n];
          float r  = xv * (1.f - g) + ov * g + se;
          outF[(size_t)m * D_MODEL + n] = r;
          outB[(size_t)m * D_MODEL + n] = f2bf(r);
        } else if constexpr (EPI == EPI_QKV) {
          float r   = v + bias[n];
          int which = n >> 10, rem = n & 1023;
          int h = rem >> 6, d = rem & 63;
          int b = m >> 10, s = m & 1023;
          size_t idx = ((((size_t)which * BATCH + b) * N_HEADS + h) * SEQ + s) * HEAD_D + d;
          outB[idx] = f2bf(r);
        } else if constexpr (EPI == EPI_BIAS) {
          outF[(size_t)m * N + n] = v + bias[n];
        } else { // EPI_GELU (exact)
          float t = v + bias[n];
          float r = 0.5f * t * (1.f + erff(t * 0.70710678118654752f));
          outB[(size_t)m * N + n] = f2bf(r);
        }
      }
    }
  }
}

// ---------------------------------------------------------------------------
// Flash attention: grid (S/128, B*H). 8 waves; each wave owns 16 query rows.
// qkv layout: [3][B*H][S][64] bf16. Online softmax, WMMA for QK^T and PV.
// ---------------------------------------------------------------------------
__global__ __launch_bounds__(256, 2)
void attn_kernel(const __bf16* __restrict__ qkv,
                 const float*  __restrict__ relb,    // (2*MAXL-1)
                 __bf16*       __restrict__ out)     // (B,S,D)
{
  constexpr int LKV = HEAD_D + 8;
  __shared__ __bf16 Ks[64 * LKV];            // K chunk, [key][d]
  __shared__ __bf16 Vt[HEAD_D * LKV];        // V chunk transposed, [d][key]
  __shared__ __bf16 Ps[8 * 16 * LKV];        // per-wave P staging, [qrow][key]

  const int tid  = threadIdx.x;
  const int lane = tid & 31;
  const int wave = tid >> 5;
  const int hi   = lane >> 4;
  const int ln   = lane & 15;
  const int bh   = blockIdx.y;               // b*16 + h
  const int q0   = blockIdx.x * 128;
  const int BH   = BATCH * N_HEADS;

  const __bf16* Qg = qkv + ((size_t)(0 * BH + bh) * SEQ) * HEAD_D;
  const __bf16* Kg = qkv + ((size_t)(1 * BH + bh) * SEQ) * HEAD_D;
  const __bf16* Vg = qkv + ((size_t)(2 * BH + bh) * SEQ) * HEAD_D;

  // Q fragments for this wave's 16 rows (A-layout, two K=32 steps over hd=64)
  v16bf aq[2];
  {
    const __bf16* p = Qg + (size_t)(q0 + wave * 16 + ln) * HEAD_D;
#pragma unroll
    for (int f = 0; f < 2; ++f) {
#pragma unroll
      for (int j = 0; j < 8; ++j) aq[f][j]     = p[f * 32 + 16 * hi + j];
#pragma unroll
      for (int j = 0; j < 8; ++j) aq[f][8 + j] = p[f * 32 + 16 * hi + 8 + j];
    }
  }

  float mrow[8], lrow[8];
  v8f o[4];
#pragma unroll
  for (int e = 0; e < 8; ++e) { mrow[e] = -3.0e38f; lrow[e] = 0.f; }
#pragma unroll
  for (int dt = 0; dt < 4; ++dt)
#pragma unroll
    for (int j = 0; j < 8; ++j) o[dt][j] = 0.f;

  for (int j0 = 0; j0 < SEQ; j0 += 64) {
    // ---- stage K and V^T ----
    {
      int d  = tid & 63;
      int r0 = tid >> 6;   // 0..3
#pragma unroll
      for (int rr = 0; rr < 16; ++rr) {
        int key = r0 + rr * 4;
        Ks[key * LKV + d] = Kg[(size_t)(j0 + key) * HEAD_D + d];
        Vt[d * LKV + key] = Vg[(size_t)(j0 + key) * HEAD_D + d];
      }
    }
    __syncthreads();

    // ---- scores: 16 rows x 64 keys (4 tiles) ----
    float p_[4][8];
#pragma unroll
    for (int ct = 0; ct < 4; ++ct) {
      v8f sc;
#pragma unroll
      for (int j = 0; j < 8; ++j) sc[j] = 0.f;
#pragma unroll
      for (int ks = 0; ks < 2; ++ks) {
        v16bf bk;
        const __bf16* pb = &Ks[(ct * 16 + ln) * LKV + ks * 32 + 16 * hi];
#pragma unroll
        for (int j = 0; j < 16; ++j) bk[j] = pb[j];
        sc = __builtin_amdgcn_wmma_f32_16x16x32_bf16(
            false, aq[ks], false, bk, (short)0, sc, false, false);
      }
#pragma unroll
      for (int e = 0; e < 8; ++e) {
        int qg = q0 + wave * 16 + e + 8 * hi;
        int kg = j0 + ct * 16 + ln;
        int rp = qg - kg + (SEQ - 1);
        rp = rp < 0 ? 0 : (rp > 2 * SEQ - 2 ? 2 * SEQ - 2 : rp);
        p_[ct][e] = sc[e] * 0.125f + relb[rp];   // hd^-0.5 = 1/8
      }
    }

    // ---- online softmax (rows replicated across 16-lane groups) ----
#pragma unroll
    for (int e = 0; e < 8; ++e) {
      float mx = p_[0][e];
#pragma unroll
      for (int ct = 1; ct < 4; ++ct) mx = fmaxf(mx, p_[ct][e]);
#pragma unroll
      for (int off = 8; off >= 1; off >>= 1) mx = fmaxf(mx, __shfl_xor(mx, off, 32));
      float mn    = fmaxf(mrow[e], mx);
      float alpha = __expf(mrow[e] - mn);
      float rs    = 0.f;
#pragma unroll
      for (int ct = 0; ct < 4; ++ct) { p_[ct][e] = __expf(p_[ct][e] - mn); rs += p_[ct][e]; }
#pragma unroll
      for (int off = 8; off >= 1; off >>= 1) rs += __shfl_xor(rs, off, 32);
      lrow[e] = lrow[e] * alpha + rs;
      mrow[e] = mn;
#pragma unroll
      for (int dt = 0; dt < 4; ++dt) o[dt][e] *= alpha;
    }

    // ---- stash P (C-layout -> LDS), then PV via WMMA ----
    __bf16* Pw = &Ps[wave * 16 * LKV];
#pragma unroll
    for (int ct = 0; ct < 4; ++ct)
#pragma unroll
      for (int e = 0; e < 8; ++e)
        Pw[(e + 8 * hi) * LKV + ct * 16 + ln] = f2bf(p_[ct][e]);

#pragma unroll
    for (int ks = 0; ks < 2; ++ks) {
      v16bf ap;
      const __bf16* pp = &Pw[ln * LKV + ks * 32 + 16 * hi];
#pragma unroll
      for (int j = 0; j < 8; ++j) ap[j]     = pp[j];
#pragma unroll
      for (int j = 0; j < 8; ++j) ap[8 + j] = pp[8 + j];
#pragma unroll
      for (int dt = 0; dt < 4; ++dt) {
        v16bf bv;
        const __bf16* pv = &Vt[(dt * 16 + ln) * LKV + ks * 32 + 16 * hi];
#pragma unroll
        for (int j = 0; j < 16; ++j) bv[j] = pv[j];
        o[dt] = __builtin_amdgcn_wmma_f32_16x16x32_bf16(
            false, ap, false, bv, (short)0, o[dt], false, false);
      }
    }
    __syncthreads();
  }

  // ---- finalize + store to (B,S,D) bf16 ----
  const int b = bh >> 4, h = bh & 15;
#pragma unroll
  for (int dt = 0; dt < 4; ++dt)
#pragma unroll
    for (int e = 0; e < 8; ++e) {
      int s    = q0 + wave * 16 + e + 8 * hi;
      int dcol = h * HEAD_D + dt * 16 + ln;
      out[((size_t)(b * SEQ + s)) * D_MODEL + dcol] = f2bf(o[dt][e] / lrow[e]);
    }
}

// ---------------------------------------------------------------------------
// Residual add + LayerNorm (fp32), one block per row of 1024.
// ---------------------------------------------------------------------------
__global__ __launch_bounds__(256, 4)
void resid_ln_kernel(const float* __restrict__ a, const float* __restrict__ r,
                     const float* __restrict__ g, const float* __restrict__ be,
                     float* __restrict__ outF, __bf16* __restrict__ outB)
{
  __shared__ float red[16];
  const int row = blockIdx.x;
  const int tid = threadIdx.x;
  const float* pa = a + (size_t)row * D_MODEL;
  const float* pr = r + (size_t)row * D_MODEL;
  float v[4];
  float s = 0.f, sq = 0.f;
#pragma unroll
  for (int i = 0; i < 4; ++i) {
    v[i] = pa[tid + i * 256] + pr[tid + i * 256];
    s  += v[i];
    sq += v[i] * v[i];
  }
#pragma unroll
  for (int off = 16; off >= 1; off >>= 1) {
    s  += __shfl_xor(s, off, 32);
    sq += __shfl_xor(sq, off, 32);
  }
  if ((tid & 31) == 0) { red[tid >> 5] = s; red[8 + (tid >> 5)] = sq; }
  __syncthreads();
  float ts = 0.f, tq = 0.f;
#pragma unroll
  for (int w = 0; w < 8; ++w) { ts += red[w]; tq += red[8 + w]; }
  const float mean = ts * (1.f / D_MODEL);
  const float var  = tq * (1.f / D_MODEL) - mean * mean;
  const float inv  = rsqrtf(var + 1e-5f);
#pragma unroll
  for (int i = 0; i < 4; ++i) {
    int c = tid + i * 256;
    float y = (v[i] - mean) * inv * g[c] + be[c];
    outF[(size_t)row * D_MODEL + c] = y;
    if (outB) outB[(size_t)row * D_MODEL + c] = f2bf(y);
  }
}

// ---------------------------------------------------------------------------
// Host launcher
// ---------------------------------------------------------------------------
extern "C" void kernel_launch(void* const* d_in, const int* in_sizes, int n_in,
                              void* d_out, int out_size, void* d_ws, size_t ws_size,
                              hipStream_t stream) {
  (void)in_sizes; (void)n_in; (void)out_size; (void)ws_size;
  const float* x      = (const float*)d_in[0];
  const float* orig   = (const float*)d_in[1];
  const int*   tstep  = (const int*)d_in[2];
  const float* sembed = (const float*)d_in[3];
  const float* Wqkv   = (const float*)d_in[4];
  const float* bqkv   = (const float*)d_in[5];
  const float* Wout   = (const float*)d_in[6];
  const float* bout   = (const float*)d_in[7];
  const float* Wg     = (const float*)d_in[8];
  const float* bg     = (const float*)d_in[9];
  const float* relb   = (const float*)d_in[10];
  const float* W1     = (const float*)d_in[11];
  const float* b1     = (const float*)d_in[12];
  const float* W2     = (const float*)d_in[13];
  const float* b2     = (const float*)d_in[14];
  const float* ln1g   = (const float*)d_in[15];
  const float* ln1b   = (const float*)d_in[16];
  const float* ln2g   = (const float*)d_in[17];
  const float* ln2b   = (const float*)d_in[18];
  float* out = (float*)d_out;

  // ---- workspace layout (bytes) ----
  char* ws = (char*)d_ws;
  // region A (32MB): qkv (24MB) + attn (8MB); later reused as h_bf16 (32MB)
  __bf16* qkv_b   = (__bf16*)(ws + 0);
  __bf16* attn_b  = (__bf16*)(ws + 25165824);
  __bf16* h_b     = (__bf16*)(ws + 0);            // alias (after attn consumed)
  // region B (16MB): xcat; later reused as ffn_f32
  __bf16* xcat_b  = (__bf16*)(ws + 33554432);
  float*  ffn_f   = (float*) (ws + 33554432);     // alias (after xcat consumed)
  __bf16* Wg_b    = (__bf16*)(ws + 50331648);     // all weights stored N x K
  __bf16* Wqkv_b  = (__bf16*)(ws + 54525952);
  __bf16* Wout_b  = (__bf16*)(ws + 60817408);
  __bf16* W1_b    = (__bf16*)(ws + 62914560);
  __bf16* W2_b    = (__bf16*)(ws + 71303168);
  float*  xg_f    = (float*) (ws + 79691776);
  __bf16* xg_b    = (__bf16*)(ws + 96468992);
  float*  proj_f  = (float*) (ws + 104857600);
  float*  ln1_f   = (float*) (ws + 121634816);
  __bf16* ln1_b16 = (__bf16*)(ws + 138412032);

  const int TB = 256;
  auto nblk = [](int n) { return (n + 255) / 256; };

  // 1) bf16 conversions (weights transposed to N x K)
  pack_cat_kernel<<<nblk(ROWS * 2 * D_MODEL), TB, 0, stream>>>(x, orig, xcat_b);
  cvt_bf16_t_kernel<<<nblk(2 * D_MODEL * D_MODEL), TB, 0, stream>>>(Wg,   Wg_b,   2 * D_MODEL, D_MODEL);
  cvt_bf16_t_kernel<<<nblk(D_MODEL * 3 * D_MODEL), TB, 0, stream>>>(Wqkv, Wqkv_b, D_MODEL, 3 * D_MODEL);
  cvt_bf16_t_kernel<<<nblk(D_MODEL * D_MODEL), TB, 0, stream>>>(Wout, Wout_b, D_MODEL, D_MODEL);
  cvt_bf16_t_kernel<<<nblk(D_MODEL * DFF), TB, 0, stream>>>(W1, W1_b, D_MODEL, DFF);
  cvt_bf16_t_kernel<<<nblk(DFF * D_MODEL), TB, 0, stream>>>(W2, W2_b, DFF, D_MODEL);

  // 2) gate GEMM: (4096,2048)@(2048,1024) -> gated x + step embed
  gemm_bf16_kernel<EPI_GATE><<<dim3(D_MODEL / BN, ROWS / BM), TB, 0, stream>>>(
      xcat_b, Wg_b, ROWS, D_MODEL, 2 * D_MODEL, bg, xg_f, xg_b, x, orig, sembed, tstep);

  // 3) QKV GEMM: (4096,1024)@(1024,3072) -> q,k,v bf16 [3][B*H][S][64]
  gemm_bf16_kernel<EPI_QKV><<<dim3(3 * D_MODEL / BN, ROWS / BM), TB, 0, stream>>>(
      xg_b, Wqkv_b, ROWS, 3 * D_MODEL, D_MODEL, bqkv, nullptr, qkv_b,
      nullptr, nullptr, nullptr, nullptr);

  // 4) flash attention -> attn bf16 (B,S,D)
  attn_kernel<<<dim3(SEQ / 128, BATCH * N_HEADS), TB, 0, stream>>>(qkv_b, relb, attn_b);

  // 5) output projection: (4096,1024)@(1024,1024) -> proj f32
  gemm_bf16_kernel<EPI_BIAS><<<dim3(D_MODEL / BN, ROWS / BM), TB, 0, stream>>>(
      attn_b, Wout_b, ROWS, D_MODEL, D_MODEL, bout, proj_f, nullptr,
      nullptr, nullptr, nullptr, nullptr);

  // 6) LN1(xg + proj)
  resid_ln_kernel<<<ROWS, TB, 0, stream>>>(xg_f, proj_f, ln1g, ln1b, ln1_f, ln1_b16);

  // 7) FFN1: (4096,1024)@(1024,4096), exact GELU -> h bf16
  gemm_bf16_kernel<EPI_GELU><<<dim3(DFF / BN, ROWS / BM), TB, 0, stream>>>(
      ln1_b16, W1_b, ROWS, DFF, D_MODEL, b1, nullptr, h_b,
      nullptr, nullptr, nullptr, nullptr);

  // 8) FFN2: (4096,4096)@(4096,1024) -> ffn f32
  gemm_bf16_kernel<EPI_BIAS><<<dim3(D_MODEL / BN, ROWS / BM), TB, 0, stream>>>(
      h_b, W2_b, ROWS, D_MODEL, DFF, b2, ffn_f, nullptr,
      nullptr, nullptr, nullptr, nullptr);

  // 9) LN2(ln1 + ffn) -> final output (f32)
  resid_ln_kernel<<<ROWS, TB, 0, stream>>>(ln1_f, ffn_f, ln2g, ln2b, out, nullptr);
}
